// CILSTM_55009941128001
// MI455X (gfx1250) — matrix operand order repaired and checked by reading, not compiled, not run on hardware
//
#include <hip/hip_runtime.h>
#include <stdint.h>
#include <stddef.h>

// ---------------- problem constants ----------------
#define CT 512          // T
#define CB 32           // B
#define CH 768          // H
#define CNH 12          // heads
#define CDH 64          // head dim
#define CNQKV (3*CH)    // 2304
#define CNG   (4*CH)    // 3072 gate cols
#define CKG   (2*CH)    // 1536 gate K
#define NWG 24
#define TPB 256

typedef __bf16 bf16_t;
typedef bf16_t v8bf  __attribute__((ext_vector_type(8)));
typedef bf16_t v16bf __attribute__((ext_vector_type(16)));
typedef float  v8f   __attribute__((ext_vector_type(8)));

// ---------------- workspace layout (bytes, all 256B aligned) ----------------
static constexpr size_t OFF_INPUTS = 0;                                  // f32 [T][B][H] mutable copy of x
static constexpr size_t SZ_INPUTS  = (size_t)CT * CB * CH * 4;
static constexpr size_t OFF_WQKV   = OFF_INPUTS + SZ_INPUTS;             // bf16 [2304(N)][768(K)]  (Wq;Wk;Wv transposed)
static constexpr size_t SZ_WQKV    = (size_t)CNQKV * CH * 2;
static constexpr size_t OFF_WGATE  = OFF_WQKV + SZ_WQKV;                 // bf16 [3072(N)][1536(K)] ([W_ih;W_hh] transposed)
static constexpr size_t SZ_WGATE   = (size_t)CNG * CKG * 2;
static constexpr size_t OFF_BQKV   = OFF_WGATE + SZ_WGATE;               // f32 [2304]
static constexpr size_t SZ_BQKV    = (size_t)CNQKV * 4;
static constexpr size_t OFF_BG     = OFF_BQKV + SZ_BQKV;                 // f32 [3072] (b_ih + b_hh)
static constexpr size_t SZ_BG      = (size_t)CNG * 4;
static constexpr size_t OFF_ROWBF  = OFF_BG + SZ_BG;                     // bf16 [32][768] row (A operand, K<768)
static constexpr size_t SZ_ROWBF   = (size_t)CB * CH * 2;
static constexpr size_t OFF_HBUF   = OFF_ROWBF + SZ_ROWBF;               // bf16 2x[32][768] double-buffered h
static constexpr size_t SZ_HBUF    = (size_t)2 * CB * CH * 2;
static constexpr size_t OFF_ROWF   = OFF_HBUF + SZ_HBUF;                 // f32 [32][768] exact row for inputs[t]
static constexpr size_t SZ_ROWF    = (size_t)CB * CH * 4;
static constexpr size_t OFF_CBUF   = OFF_ROWF + SZ_ROWF;                 // f32 [32][768] cell state
static constexpr size_t SZ_CBUF    = (size_t)CB * CH * 4;
static constexpr size_t OFF_ASEQ   = OFF_CBUF + SZ_CBUF;                 // bf16 per-head-WG [12][64][768]
static constexpr size_t SZ_ASEQ    = (size_t)CNH * 64 * CH * 2;
static constexpr size_t OFF_BAR    = OFF_ASEQ + SZ_ASEQ;                 // unsigned[2]: count, generation

// ---------------- helpers ----------------
__device__ __forceinline__ uint16_t f2bf(float x){
  uint32_t u = __float_as_uint(x);
  return (uint16_t)((u + 0x7fffu + ((u >> 16) & 1u)) >> 16);   // RNE
}
__device__ __forceinline__ float sigm(float x){ return 1.0f / (1.0f + __expf(-x)); }

// A fragment (16x32 bf16): lane half 'hi' holds K chunks [k0+8*hi, +8) and [k0+16+8*hi, +8)
__device__ __forceinline__ v16bf load_a_frag(const uint16_t* Arow, int k0, int hi){
  v16bf a;
  reinterpret_cast<v8bf*>(&a)[0] = *reinterpret_cast<const v8bf*>(Arow + k0 + hi * 8);
  reinterpret_cast<v8bf*>(&a)[1] = *reinterpret_cast<const v8bf*>(Arow + k0 + 16 + hi * 8);
  return a;
}
// B fragment (32x16 bf16, weights stored N-major): lane half 'hi' holds K chunks [k0+8*hi, +8) and [k0+16+8*hi, +8)
__device__ __forceinline__ v16bf load_b_frag(const uint16_t* Bcol, int k0, int hi){
  v16bf b;
  const uint16_t* p = Bcol + k0 + hi * 8;
  reinterpret_cast<v8bf*>(&b)[0] = *reinterpret_cast<const v8bf*>(p);
  reinterpret_cast<v8bf*>(&b)[1] = *reinterpret_cast<const v8bf*>(p + 16);
  return b;
}

__device__ __forceinline__ void grid_sync(unsigned* cnt, unsigned* gen){
  __threadfence();
  __syncthreads();
  if (threadIdx.x == 0){
    unsigned g = __hip_atomic_load(gen, __ATOMIC_RELAXED, __HIP_MEMORY_SCOPE_AGENT);
    unsigned prev = __hip_atomic_fetch_add(cnt, 1u, __ATOMIC_ACQ_REL, __HIP_MEMORY_SCOPE_AGENT);
    if (prev == (unsigned)(NWG - 1)){
      __hip_atomic_store(cnt, 0u, __ATOMIC_RELAXED, __HIP_MEMORY_SCOPE_AGENT);
      __hip_atomic_fetch_add(gen, 1u, __ATOMIC_RELEASE, __HIP_MEMORY_SCOPE_AGENT);
    } else {
      while (__hip_atomic_load(gen, __ATOMIC_ACQUIRE, __HIP_MEMORY_SCOPE_AGENT) == g){
        __builtin_amdgcn_s_sleep(2);
      }
    }
  }
  __syncthreads();
}

// ---------------- one-time pack / init kernels ----------------
__global__ void pack_qkv_kernel(const float* wq, const float* wk, const float* wv,
                                const float* bq, const float* bk, const float* bv,
                                uint8_t* ws){
  uint16_t* wqkv = (uint16_t*)(ws + OFF_WQKV);
  float* bqkv = (float*)(ws + OFF_BQKV);
  int tid0 = blockIdx.x * blockDim.x + threadIdx.x;
  int stride = gridDim.x * blockDim.x;
  for (int idx = tid0; idx < CNQKV * CH; idx += stride){
    int n = idx / CH, k = idx % CH;
    int sec = n / CH, nl = n % CH;
    const float* w = (sec == 0) ? wq : (sec == 1) ? wk : wv;     // (H,H) row-major: w[k][nl]
    wqkv[(size_t)n * CH + k] = f2bf(w[(size_t)k * CH + nl]);
  }
  for (int n = tid0; n < CNQKV; n += stride){
    int sec = n / CH, nl = n % CH;
    bqkv[n] = (sec == 0) ? bq[nl] : (sec == 1) ? bk[nl] : bv[nl];
  }
}

__global__ void pack_gates_kernel(const float* wih, const float* whh,
                                  const float* bih, const float* bhh,
                                  uint8_t* ws){
  uint16_t* wg = (uint16_t*)(ws + OFF_WGATE);
  float* bg = (float*)(ws + OFF_BG);
  int tid0 = blockIdx.x * blockDim.x + threadIdx.x;
  int stride = gridDim.x * blockDim.x;
  for (long long idx = tid0; idx < (long long)CNG * CKG; idx += stride){
    int n = (int)(idx / CKG), k = (int)(idx % CKG);
    float v = (k < CH) ? wih[(size_t)k * CNG + n] : whh[(size_t)(k - CH) * CNG + n];
    wg[(size_t)n * CKG + k] = f2bf(v);
  }
  for (int n = tid0; n < CNG; n += stride) bg[n] = bih[n] + bhh[n];
}

__global__ void init_state_kernel(const float* h0, const float* c0, uint8_t* ws){
  float* cbuf = (float*)(ws + OFF_CBUF);
  uint16_t* hbuf = (uint16_t*)(ws + OFF_HBUF);      // buffer 0 is read at t=0
  unsigned* bar = (unsigned*)(ws + OFF_BAR);
  int tid0 = blockIdx.x * blockDim.x + threadIdx.x;
  int stride = gridDim.x * blockDim.x;
  for (int idx = tid0; idx < CB * CH; idx += stride){
    cbuf[idx] = c0[idx];
    hbuf[idx] = f2bf(h0[idx]);
  }
  if (tid0 < 2) bar[tid0] = 0u;
}

// ---------------- persistent scan kernel ----------------
__global__ __launch_bounds__(TPB) void cilstm_main_kernel(const int* __restrict__ ping,
                                                          float* __restrict__ out,
                                                          uint8_t* __restrict__ ws){
  float*          inputs = (float*)(ws + OFF_INPUTS);
  const uint16_t* wqkv   = (const uint16_t*)(ws + OFF_WQKV);
  const uint16_t* wgate  = (const uint16_t*)(ws + OFF_WGATE);
  const float*    bqkv   = (const float*)(ws + OFF_BQKV);
  const float*    bg     = (const float*)(ws + OFF_BG);
  uint16_t*       rowbf  = (uint16_t*)(ws + OFF_ROWBF);
  uint16_t*       hbuf   = (uint16_t*)(ws + OFF_HBUF);
  float*          rowf   = (float*)(ws + OFF_ROWF);
  float*          cbuf   = (float*)(ws + OFF_CBUF);
  uint16_t*       aseqA  = (uint16_t*)(ws + OFF_ASEQ);
  unsigned*       bar    = (unsigned*)(ws + OFF_BAR);

  __shared__ float smem[64 * 192];   // phase1: qkv [64][192]; phase2: gates [32][128]

  const int wg   = blockIdx.x;
  const int tid  = threadIdx.x;
  const int wave = tid >> 5;
  const int lane = tid & 31;
  const int l15  = lane & 15;
  const int hi   = lane >> 4;

  for (int t = 0; t < CT; ++t){
    const uint16_t* hcur  = hbuf + (size_t)(t & 1) * CB * CH;
    uint16_t*       hnext = hbuf + (size_t)((t + 1) & 1) * CB * CH;

    // ================= phase 1: attention (WGs 0..11, one head each) =================
    if (wg < CNH){
      const int head = wg;
      uint16_t* aseq = aseqA + (size_t)wg * 64 * CH;

      // Build A = [left(32) ; right(32)] rows in bf16 (redundant per head-WG: saves a grid sync)
      for (int q4 = tid; q4 < (64 * CH) / 4; q4 += TPB){
        int e = q4 * 4;
        int r = e / CH, cc = e % CH;
        const float* src;
        if (r < 32){
          int p = ping[r * CT + t];
          int s = p - 1; s = (s < 0) ? 0 : ((s > CT - 1) ? (CT - 1) : s);
          src = inputs + (size_t)s * CB * CH + (size_t)r * CH + cc;
        } else {
          src = inputs + (size_t)t * CB * CH + (size_t)(r - 32) * CH + cc;
        }
        float4 v4 = *reinterpret_cast<const float4*>(src);
        ushort4 o4;
        o4.x = f2bf(v4.x); o4.y = f2bf(v4.y); o4.z = f2bf(v4.z); o4.w = f2bf(v4.w);
        *reinterpret_cast<ushort4*>(aseq + e) = o4;
      }
      __threadfence_block();
      __syncthreads();

      // q|k|v GEMM for this head: (64x768) x (768x192) -> smem.
      // 24 units of (m-pair, n-tile): two independent accumulators share one B fragment,
      // giving the XDL pipe two chains and cutting fragment loads to 3 per tile.
      for (int i = 0; i < 3; ++i){
        int unit = wave + 8 * i;                 // 0..23
        int mp = unit >> 4 ? 1 : (unit >> 3) & 1; // unit/12? keep simple below
        // map: unit = mp*12 + nt  (mp in {0,1}, nt in 0..11)
        mp = unit / 12;
        int nt = unit - mp * 12;
        int sec = nt >> 2, ntl = nt & 3;         // sec: 0=q,1=k,2=v
        int n0g = sec * CH + head * CDH + ntl * 16;
        int m0a = mp * 32, m0b = mp * 32 + 16;
        v8f acc0 = {0.f,0.f,0.f,0.f,0.f,0.f,0.f,0.f};
        v8f acc1 = {0.f,0.f,0.f,0.f,0.f,0.f,0.f,0.f};
        const uint16_t* ArowA = aseq + (size_t)(m0a + l15) * CH;
        const uint16_t* ArowB = aseq + (size_t)(m0b + l15) * CH;
        const uint16_t* Bcol  = wqkv + (size_t)(n0g + l15) * CH;
        __builtin_prefetch(Bcol, 0, 1);
        __builtin_prefetch(Bcol + CH - 64, 0, 1);
        #pragma unroll 4
        for (int k0 = 0; k0 < CH; k0 += 32){
          v16bf b  = load_b_frag(Bcol, k0, hi);
          v16bf a0 = load_a_frag(ArowA, k0, hi);
          v16bf a1 = load_a_frag(ArowB, k0, hi);
          acc0 = __builtin_amdgcn_wmma_f32_16x16x32_bf16(false, a0, false, b,
                                                         (short)0, acc0, false, false);
          acc1 = __builtin_amdgcn_wmma_f32_16x16x32_bf16(false, a1, false, b,
                                                         (short)0, acc1, false, false);
        }
        int cl = sec * 64 + ntl * 16 + l15;
        float bias = bqkv[n0g + l15];
        #pragma unroll
        for (int r = 0; r < 8; ++r){
          smem[(size_t)(m0a + r + hi * 8) * 192 + cl] = acc0[r] + bias;
          smem[(size_t)(m0b + r + hi * 8) * 192 + cl] = acc1[r] + bias;
        }
      }
      __syncthreads();

      // 2-token softmax attention, 8 lanes per batch element, shuffle-tree reduction.
      {
        int b   = wave * 4 + (lane >> 3);        // 0..31
        int sub = lane & 7;                      // 8 lanes per b, 8 dh elems each
        const float* qv = &smem[(size_t)(32 + b) * 192 + 0];
        const float* kL = &smem[(size_t)b * 192 + 64];
        const float* kR = &smem[(size_t)(32 + b) * 192 + 64];
        const float* vL = &smem[(size_t)b * 192 + 128];
        const float* vR = &smem[(size_t)(32 + b) * 192 + 128];
        float s0 = 0.f, s1 = 0.f;
        #pragma unroll
        for (int j = 0; j < 8; ++j){
          int i = sub * 8 + j;
          s0 += qv[i] * kL[i];
          s1 += qv[i] * kR[i];
        }
        #pragma unroll
        for (int off = 1; off < 8; off <<= 1){
          s0 += __shfl_xor(s0, off, 8);
          s1 += __shfl_xor(s1, off, 8);
        }
        s0 *= 0.125f; s1 *= 0.125f;              // 1/sqrt(dh)
        float m = fmaxf(s0, s1);
        float e0 = __expf(s0 - m), e1 = __expf(s1 - m);
        float inv = 1.f / (e0 + e1);
        e0 *= inv; e1 *= inv;
        int p = ping[b * CT + t];
        bool mask = (p != 0) && (t > 0);
        const float* rightrow = inputs + (size_t)t * CB * CH + (size_t)b * CH + head * CDH;
        #pragma unroll
        for (int j = 0; j < 8; ++j){
          int i = sub * 8 + j;
          float ctx = e0 * vL[i] + e1 * vR[i];
          float rv = mask ? ctx : rightrow[i];
          rowf [(size_t)b * CH + head * CDH + i] = rv;
          rowbf[(size_t)b * CH + head * CDH + i] = f2bf(rv);
        }
      }
    }

    grid_sync(&bar[0], &bar[1]);   // row visible everywhere; all inputs reads done

    // ================= phase 2: gate GEMM + LSTM (all WGs; 32 hidden units each) =====
    // Commit inputs[t] = row (deferred past the barrier -> phase-1 readers saw old buffer)
    for (int idx = tid; idx < CB * 32; idx += TPB){
      int b = idx >> 5, u = idx & 31;
      int n = wg * 32 + u;
      inputs[(size_t)t * CB * CH + (size_t)b * CH + n] = rowf[(size_t)b * CH + n];
    }

    // gates = [row | h_prev] (32x1536) x Wg^T slice (1536x128) -> smem[32][128].
    // One n-tile per wave, both m-tiles in flight (independent accumulators, shared B).
    {
      int nt = wave;                             // 0..7
      int sec = nt >> 1, ntl = nt & 1;           // sec: 0=i,1=f,2=g,3=o
      int n0g = sec * CH + wg * 32 + ntl * 16;
      v8f acc0 = {0.f,0.f,0.f,0.f,0.f,0.f,0.f,0.f};
      v8f acc1 = {0.f,0.f,0.f,0.f,0.f,0.f,0.f,0.f};
      const uint16_t* ArowR0 = rowbf + (size_t)(0  + l15) * CH;
      const uint16_t* ArowR1 = rowbf + (size_t)(16 + l15) * CH;
      const uint16_t* ArowH0 = hcur  + (size_t)(0  + l15) * CH;
      const uint16_t* ArowH1 = hcur  + (size_t)(16 + l15) * CH;
      const uint16_t* Bcol   = wgate + (size_t)(n0g + l15) * CKG;
      __builtin_prefetch(Bcol, 0, 1);
      __builtin_prefetch(Bcol + CKG - 64, 0, 1);
      #pragma unroll 4
      for (int k0 = 0; k0 < CH; k0 += 32){       // K half 1: row operand
        v16bf b  = load_b_frag(Bcol, k0, hi);
        v16bf a0 = load_a_frag(ArowR0, k0, hi);
        v16bf a1 = load_a_frag(ArowR1, k0, hi);
        acc0 = __builtin_amdgcn_wmma_f32_16x16x32_bf16(false, a0, false, b,
                                                       (short)0, acc0, false, false);
        acc1 = __builtin_amdgcn_wmma_f32_16x16x32_bf16(false, a1, false, b,
                                                       (short)0, acc1, false, false);
      }
      #pragma unroll 4
      for (int k0 = 0; k0 < CH; k0 += 32){       // K half 2: h_prev operand
        v16bf b  = load_b_frag(Bcol, CH + k0, hi);
        v16bf a0 = load_a_frag(ArowH0, k0, hi);
        v16bf a1 = load_a_frag(ArowH1, k0, hi);
        acc0 = __builtin_amdgcn_wmma_f32_16x16x32_bf16(false, a0, false, b,
                                                       (short)0, acc0, false, false);
        acc1 = __builtin_amdgcn_wmma_f32_16x16x32_bf16(false, a1, false, b,
                                                       (short)0, acc1, false, false);
      }
      int cl = sec * 32 + ntl * 16 + l15;
      float bias = bg[n0g + l15];
      #pragma unroll
      for (int r = 0; r < 8; ++r){
        smem[(size_t)(0  + r + hi * 8) * 128 + cl] = acc0[r] + bias;
        smem[(size_t)(16 + r + hi * 8) * 128 + cl] = acc1[r] + bias;
      }
    }
    __syncthreads();

    // LSTM elementwise: all four gates of a unit live in this WG's smem slice
    for (int idx = tid; idx < CB * 32; idx += TPB){
      int b = idx >> 5, u = idx & 31;
      int n = wg * 32 + u;
      float ig = smem[(size_t)b * 128 + u];
      float fg = smem[(size_t)b * 128 + 32 + u];
      float gg = smem[(size_t)b * 128 + 64 + u];
      float og = smem[(size_t)b * 128 + 96 + u];
      float c = sigm(fg) * cbuf[(size_t)b * CH + n] + sigm(ig) * tanhf(gg);
      float h = sigm(og) * tanhf(c);
      cbuf[(size_t)b * CH + n] = c;
      out[(size_t)t * CB * CH + (size_t)b * CH + n] = h;     // hs[t]
      hnext[(size_t)b * CH + n] = f2bf(h);
      if (t == CT - 1){
        out[(size_t)CT * CB * CH + (size_t)b * CH + n] = h;                      // hT
        out[(size_t)CT * CB * CH + (size_t)CB * CH + (size_t)b * CH + n] = c;    // cT
      }
    }

    grid_sync(&bar[0], &bar[1]);   // h visible; safe to overwrite row / aseq next step
  }
}

// ---------------- host entry ----------------
extern "C" void kernel_launch(void* const* d_in, const int* in_sizes, int n_in,
                              void* d_out, int out_size, void* d_ws, size_t ws_size,
                              hipStream_t stream){
  (void)in_sizes; (void)n_in; (void)out_size; (void)ws_size;
  const float* x    = (const float*)d_in[0];
  const int*   ping = (const int*)  d_in[1];
  const float* h0   = (const float*)d_in[2];
  const float* c0   = (const float*)d_in[3];
  const float* wih  = (const float*)d_in[4];
  const float* whh  = (const float*)d_in[5];
  const float* bih  = (const float*)d_in[6];
  const float* bhh  = (const float*)d_in[7];
  const float* wq   = (const float*)d_in[8];
  const float* bq   = (const float*)d_in[9];
  const float* wk   = (const float*)d_in[10];
  const float* bk   = (const float*)d_in[11];
  const float* wv   = (const float*)d_in[12];
  const float* bv   = (const float*)d_in[13];
  uint8_t* ws = (uint8_t*)d_ws;

  // Mutable copy of x (the scan rewrites inputs[t] each step)
  hipMemcpyAsync(ws + OFF_INPUTS, x, SZ_INPUTS, hipMemcpyDeviceToDevice, stream);

  // One-time weight packing to bf16, transposed N-major for contiguous B fragments
  pack_qkv_kernel  <<<512,  TPB, 0, stream>>>(wq, wk, wv, bq, bk, bv, ws);
  pack_gates_kernel<<<1024, TPB, 0, stream>>>(wih, whh, bih, bhh, ws);
  init_state_kernel<<<96,   TPB, 0, stream>>>(h0, c0, ws);

  // Persistent scan: 24 resident workgroups, 2 grid syncs per time step
  cilstm_main_kernel<<<NWG, TPB, 0, stream>>>(ping, (float*)d_out, ws);
}
